// Attention_33122787787565
// MI455X (gfx1250) — matrix operand order repaired and checked by reading, not compiled, not run on hardware
//
#include <hip/hip_runtime.h>
#include <hip/hip_bf16.h>

// Problem constants
#define BB   4
#define NN   2048
#define DD   768
#define HH   12
#define DHH  64
#define BN   (BB * NN)      // 8192 tokens total
#define EPSF 1e-6f

typedef __attribute__((ext_vector_type(16))) __bf16 bf16x16;
typedef __attribute__((ext_vector_type(8)))  __bf16 bf16x8;
typedef __attribute__((ext_vector_type(8)))  float  f32x8;
typedef __attribute__((ext_vector_type(4)))  float  f32x4;

// ---------------------------------------------------------------------------
// WMMA fragment helpers (layouts per CDNA5 ISA 7.12.2, wave32)
// ---------------------------------------------------------------------------

static __device__ __forceinline__ f32x8 wmma_bf16(bf16x16 a, bf16x16 b, f32x8 c) {
  return __builtin_amdgcn_wmma_f32_16x16x32_bf16(
      /*neg_a=*/false, a, /*neg_b=*/false, b,
      /*c_mod=*/(short)0, c, /*reuse_a=*/false, /*reuse_b=*/false);
}

// A fragment from bf16 row-major (global or LDS); 2x 16B loads
static __device__ __forceinline__ bf16x16
frag_A_bf16(const __bf16* __restrict__ X, int lda, int lane) {
  const __bf16* rp = X + (size_t)(lane & 15) * lda + ((lane >> 4) << 3);
  bf16x8 lo = *(const bf16x8*)rp;
  bf16x8 hi = *(const bf16x8*)(rp + 16);
  return __builtin_shufflevector(lo, hi, 0, 1, 2, 3, 4, 5, 6, 7,
                                 8, 9, 10, 11, 12, 13, 14, 15);
}

// A fragment from f32 row-major (LDS probs strip); converts to bf16
static __device__ __forceinline__ bf16x16
frag_A_f32(const float* __restrict__ X, int lda, int lane) {
  const float* rp = X + (size_t)(lane & 15) * lda + ((lane >> 4) << 3);
  f32x4 q0 = *(const f32x4*)(rp);
  f32x4 q1 = *(const f32x4*)(rp + 4);
  f32x4 q2 = *(const f32x4*)(rp + 16);
  f32x4 q3 = *(const f32x4*)(rp + 20);
  bf16x16 a;
#pragma unroll
  for (int e = 0; e < 4; ++e) {
    a[e]      = (__bf16)q0[e];
    a[4 + e]  = (__bf16)q1[e];
    a[8 + e]  = (__bf16)q2[e];
    a[12 + e] = (__bf16)q3[e];
  }
  return a;
}

// B fragment = transposed chunk: B[k][n] = M[n][k]; M bf16 row-major at (n0,k0)
static __device__ __forceinline__ bf16x16
frag_BT_bf16(const __bf16* __restrict__ M, int ldm, int lane) {
  const __bf16* rp = M + (size_t)(lane & 15) * ldm + ((lane >> 4) << 3);
  bf16x8 lo = *(const bf16x8*)rp;
  bf16x8 hi = *(const bf16x8*)(rp + 16);
  return __builtin_shufflevector(lo, hi, 0, 1, 2, 3, 4, 5, 6, 7,
                                 8, 9, 10, 11, 12, 13, 14, 15);
}

// Convert two f32x4 into one bf16x8
static __device__ __forceinline__ bf16x8 cvt_bf16x8(f32x4 a, f32x4 b) {
  bf16x8 r;
#pragma unroll
  for (int e = 0; e < 4; ++e) {
    r[e]     = (__bf16)a[e];
    r[4 + e] = (__bf16)b[e];
  }
  return r;
}

// gfx1250 async global->LDS 16B copy (ASYNCcnt-tracked) + wait
static __device__ __forceinline__ void async_copy_b128(const __bf16* gptr,
                                                       __bf16* lptr) {
  const unsigned lds = (unsigned)(uintptr_t)lptr;           // LDS address
  const unsigned long long ga = (unsigned long long)(uintptr_t)gptr;
  asm volatile("global_load_async_to_lds_b128 %0, %1, off"
               :: "v"(lds), "v"(ga) : "memory");
}
static __device__ __forceinline__ void wait_asynccnt0() {
  asm volatile("s_wait_asynccnt 0x0" ::: "memory");
}

// ---------------------------------------------------------------------------
// Kernel 1: LDS-tiled projection GEMM, ping-pong panels (one barrier/step).
// 256 threads (8 waves) compute a 64x64 tile of X(8192x768) @ W^T + bias.
// TV==0: dst (B,H,N,DH) for Q,K;  TV==1: dst (B,H,DH,N) for V.
// ---------------------------------------------------------------------------
template <int TV>
__global__ __launch_bounds__(256) void proj_tiled_kernel(
    const float* __restrict__ X, const float* __restrict__ W,
    const float* __restrict__ bias, __bf16* __restrict__ dst, float scale) {
  __shared__ __bf16 Al[2][64 * 32];
  __shared__ __bf16 Bl[2][64 * 32];
  const int tid  = threadIdx.x;
  const int lane = tid & 31;
  const int wv   = tid >> 5;                 // 0..7
  const int ct = blockIdx.x % (DD / 64);     // 12
  const int rt = blockIdx.x / (DD / 64);     // 128
  const int m0 = rt << 6, n0 = ct << 6;

  const int pr = tid >> 2;                   // panel row 0..63
  const int pk = (tid & 3) << 3;             // panel k 0,8,16,24

  const int wm = (wv >> 1) << 4;             // wave row offset 0..48
  const int wn = (wv & 1) << 5;              // wave col offset 0,32

  const float* xp = X + (size_t)(m0 + pr) * DD + pk;
  const float* wp = W + (size_t)(n0 + pr) * DD + pk;

  // preload panel 0
  *(bf16x8*)(Al[0] + pr * 32 + pk) =
      cvt_bf16x8(*(const f32x4*)xp, *(const f32x4*)(xp + 4));
  *(bf16x8*)(Bl[0] + pr * 32 + pk) =
      cvt_bf16x8(*(const f32x4*)wp, *(const f32x4*)(wp + 4));
  __syncthreads();

  f32x8 acc0 = {}, acc1 = {};
  const int NSTEP = DD / 32;                 // 24
  for (int ks = 0; ks < NSTEP; ++ks) {
    const int cur = ks & 1;
    const bool have = (ks + 1) < NSTEP;
    f32x4 na0 = {}, na1 = {}, nb0 = {}, nb1 = {};
    if (have) {                              // issue next panel's loads early
      const int k1 = (ks + 1) << 5;
      na0 = *(const f32x4*)(xp + k1);
      na1 = *(const f32x4*)(xp + k1 + 4);
      nb0 = *(const f32x4*)(wp + k1);
      nb1 = *(const f32x4*)(wp + k1 + 4);
    }
    bf16x16 a = frag_A_bf16(Al[cur] + wm * 32, 32, lane);
    acc0 = wmma_bf16(a, frag_BT_bf16(Bl[cur] + wn * 32, 32, lane), acc0);
    acc1 = wmma_bf16(a, frag_BT_bf16(Bl[cur] + (wn + 16) * 32, 32, lane), acc1);
    if (have) {
      *(bf16x8*)(Al[cur ^ 1] + pr * 32 + pk) = cvt_bf16x8(na0, na1);
      *(bf16x8*)(Bl[cur ^ 1] + pr * 32 + pk) = cvt_bf16x8(nb0, nb1);
    }
    __syncthreads();
  }

  const int mb = m0 + wm + ((lane >> 4) << 3);
#pragma unroll
  for (int t = 0; t < 2; ++t) {
    const f32x8 acc = t ? acc1 : acc0;
    const int n = n0 + wn + (t << 4) + (lane & 15);
    const int h = n / DHH, d = n % DHH;
    const float bvv = bias[n];
#pragma unroll
    for (int r = 0; r < 8; ++r) {
      const int m = mb + r;
      const int bidx = m >> 11, tok = m & (NN - 1);
      const float val = (acc[r] + bvv) * scale;
      const size_t zo = (size_t)bidx * HH + h;
      if (TV)
        dst[(zo * DHH + d) * NN + tok] = (__bf16)val;
      else
        dst[(zo * NN + tok) * DHH + d] = (__bf16)val;
    }
  }
}

// ---------------------------------------------------------------------------
// Kernel 2 (fused): per (b,h, 32-query strip), 512 threads = 16 wave32:
//   scores strip (32x2048) via WMMA -> LDS (270KB, one WG per WGP)
//   softmax-with-policy in LDS, probs written to HBM once
//   ctx strip (32x64) = P_strip @ V via WMMA reading A-frags from LDS
// Scores/ctx loops software-pipelined (next fragments loaded before WMMA).
// ---------------------------------------------------------------------------
#define STRIP        32
#define SM_STRIP_F32 (STRIP * NN)        // 65536 floats
#define SM_PACC_F32  (8 * 256)           // 2048 floats
#define SM_BYTES     ((SM_STRIP_F32 + SM_PACC_F32) * 4)   // 270336 B

__global__ __launch_bounds__(512) void attn_fused_kernel(
    const __bf16* __restrict__ Q, const __bf16* __restrict__ K,
    const __bf16* __restrict__ Vt, const float* __restrict__ policy,
    float* __restrict__ probs, __bf16* __restrict__ ctx2d) {
  extern __shared__ float lds[];
  float* Srow = lds;                     // [32][2048] score strip
  float* Pacc = lds + SM_STRIP_F32;      // 8 tiles x 16x16 partials

  const int tid  = threadIdx.x;
  const int lane = tid & 31;
  const int wv   = tid >> 5;             // wave id 0..15
  const int z    = blockIdx.z;           // b*H + h
  const int b    = z / HH;
  const int h    = z % HH;
  const int m0   = blockIdx.x * STRIP;   // query strip base

  const __bf16* Kp = K + (size_t)z * NN * DHH;

  // ---- scores: wave wv -> row-tile (wv&1), column tiles [(wv>>1)*16 ..] ----
  {
    const int rtile = wv & 1;
    const __bf16* Qp = Q + ((size_t)z * NN + m0 + rtile * 16) * DHH;
    const bf16x16 a0 = frag_A_bf16(Qp, DHH, lane);
    const bf16x16 a1 = frag_A_bf16(Qp + 32, DHH, lane);

    const int n_lane = lane & 15;
    const int rb     = (rtile << 4) + ((lane >> 4) << 3);
    const int cbase  = (wv >> 1) << 8;   // column base (16 tiles of 16)

    const __bf16* kp = Kp + (size_t)cbase * DHH;
    bf16x16 b0 = frag_BT_bf16(kp, DHH, lane);
    bf16x16 b1 = frag_BT_bf16(kp + 32, DHH, lane);
#pragma unroll 2
    for (int t = 0; t < 16; ++t) {
      bf16x16 nb0 = b0, nb1 = b1;
      if (t < 15) {                      // pipeline next tile's K fragments
        const __bf16* kpn = kp + (size_t)(t + 1) * 16 * DHH;
        if (t < 14) __builtin_prefetch(kpn + 16 * DHH, 0, 3);
        nb0 = frag_BT_bf16(kpn, DHH, lane);
        nb1 = frag_BT_bf16(kpn + 32, DHH, lane);
      }
      f32x8 acc = {};
      acc = wmma_bf16(a0, b0, acc);
      acc = wmma_bf16(a1, b1, acc);
      float* sp = Srow + (size_t)rb * NN + cbase + (t << 4) + n_lane;
#pragma unroll
      for (int r = 0; r < 8; ++r) sp[(size_t)r * NN] = acc[r];
      b0 = nb0;
      b1 = nb1;
    }
  }
  __syncthreads();

  // ---- softmax with policy: wave wv handles rows 2*wv, 2*wv+1 ----
  const float* pol = policy + (size_t)b * NN;
#pragma unroll
  for (int rr = 0; rr < 2; ++rr) {
    const int r = wv * 2 + rr;
    const int i = m0 + r;                // global query index
    float* row = Srow + (size_t)r * NN;

    float mx = -3.4e38f;
    for (int c = lane * 4; c < NN; c += 128) {
      f32x4 v = *(const f32x4*)(row + c);
      mx = fmaxf(mx, fmaxf(fmaxf(v[0], v[1]), fmaxf(v[2], v[3])));
    }
#pragma unroll
    for (int o = 16; o > 0; o >>= 1) mx = fmaxf(mx, __shfl_xor(mx, o, 32));

    float sum = 0.f;
    for (int c = lane * 4; c < NN; c += 128) {
      f32x4 v = *(const f32x4*)(row + c);
#pragma unroll
      for (int e = 0; e < 4; ++e) {
        const int j = c + e;
        const float ap = (j == i) ? 1.0f : pol[j];
        v[e] = __expf(v[e] - mx) * ap;
        sum += v[e];
      }
      *(f32x4*)(row + c) = v;
    }
#pragma unroll
    for (int o = 16; o > 0; o >>= 1) sum += __shfl_xor(sum, o, 32);

    const float inv = 1.0f / (sum + EPSF);
    const float c0  = EPSF / (float)NN;
    float* gout = probs + ((size_t)z * NN + i) * NN;
    for (int c = lane * 4; c < NN; c += 128) {
      f32x4 v = *(const f32x4*)(row + c);
#pragma unroll
      for (int e = 0; e < 4; ++e) v[e] = (v[e] + c0) * inv;
      *(f32x4*)(row + c) = v;            // keep for ctx GEMM
      *(f32x4*)(gout + c) = v;           // probs output (written once)
    }
  }
  __syncthreads();

  // ---- ctx strip: 8 tiles (2 row x 4 col) x 2 K-halves = 16 wave tasks ----
  const int rt2   = wv & 1;              // row tile 0..1
  const int ctt   = (wv >> 1) & 3;       // d-tile 0..3
  const int khalf = wv >> 3;             // K half
  const int n0    = ctt << 4;
  const __bf16* Vtp = Vt + ((size_t)z * DHH + n0) * NN;  // layout [d][tok]
  const float* Sp = Srow + (size_t)(rt2 << 4) * NN;

  f32x8 acc = {};
  const int kbeg = khalf * (NN / 2);
  const int kend = kbeg + NN / 2;
  bf16x16 bb = frag_BT_bf16(Vtp + kbeg, NN, lane);
  for (int k0 = kbeg; k0 < kend; k0 += 32) {
    bf16x16 nbb = bb;
    if (k0 + 32 < kend) nbb = frag_BT_bf16(Vtp + k0 + 32, NN, lane);
    bf16x16 a = frag_A_f32(Sp + k0, NN, lane);
    acc = wmma_bf16(a, bb, acc);
    bb = nbb;
  }

  const int n  = lane & 15;
  const int rb = (lane >> 4) << 3;
  const int tile = (rt2 << 2) + ctt;
  if (khalf == 1) {
    float* pp = Pacc + tile * 256;
#pragma unroll
    for (int r = 0; r < 8; ++r) pp[(rb + r) * 16 + n] = acc[r];
  }
  __syncthreads();
  if (khalf == 0) {
    const float* pp = Pacc + tile * 256;
#pragma unroll
    for (int r = 0; r < 8; ++r) {
      const float val = acc[r] + pp[(rb + r) * 16 + n];
      const int tok = m0 + (rt2 << 4) + rb + r;
      ctx2d[((size_t)b * NN + tok) * DD + h * DHH + n0 + n] = (__bf16)val;
    }
  }
}

// ---------------------------------------------------------------------------
// Kernel 3: LDS-tiled output projection: ctx2d(8192x768 bf16) @ Wo^T + bo.
// Ping-pong panels; A panel is a verbatim bf16 copy -> use gfx1250 async
// global->LDS b128 transfers (ASYNCcnt), B panel converts f32->bf16 via VGPRs.
// ---------------------------------------------------------------------------
__global__ __launch_bounds__(256) void out_tiled_kernel(
    const __bf16* __restrict__ ctx2d, const float* __restrict__ Wo,
    const float* __restrict__ bo, float* __restrict__ out) {
  __shared__ __bf16 Al[2][64 * 32];
  __shared__ __bf16 Bl[2][64 * 32];
  const int tid  = threadIdx.x;
  const int lane = tid & 31;
  const int wv   = tid >> 5;
  const int ct = blockIdx.x % (DD / 64);
  const int rt = blockIdx.x / (DD / 64);
  const int m0 = rt << 6, n0 = ct << 6;

  const int pr = tid >> 2;
  const int pk = (tid & 3) << 3;
  const int wm = (wv >> 1) << 4;
  const int wn = (wv & 1) << 5;

  const __bf16* ap = ctx2d + (size_t)(m0 + pr) * DD + pk;
  const float*  wp = Wo + (size_t)(n0 + pr) * DD + pk;
  __bf16* const al = &Al[0][0] + pr * 32 + pk;
  __bf16* const bl = &Bl[0][0] + pr * 32 + pk;
  const int PHASE = 64 * 32;                 // ping-pong offset (elements)

  // preload panel 0 (A async, B converted)
  async_copy_b128(ap, al);
  *(bf16x8*)bl = cvt_bf16x8(*(const f32x4*)wp, *(const f32x4*)(wp + 4));
  wait_asynccnt0();
  __syncthreads();

  f32x8 acc0 = {}, acc1 = {};
  const int NSTEP = DD / 32;                 // 24
  for (int ks = 0; ks < NSTEP; ++ks) {
    const int cur = ks & 1;
    const bool have = (ks + 1) < NSTEP;
    f32x4 nb0 = {}, nb1 = {};
    if (have) {
      const int k1 = (ks + 1) << 5;
      async_copy_b128(ap + k1, al + (cur ^ 1) * PHASE);   // A: async DMA
      nb0 = *(const f32x4*)(wp + k1);
      nb1 = *(const f32x4*)(wp + k1 + 4);
    }
    bf16x16 a = frag_A_bf16(Al[cur] + wm * 32, 32, lane);
    acc0 = wmma_bf16(a, frag_BT_bf16(Bl[cur] + wn * 32, 32, lane), acc0);
    acc1 = wmma_bf16(a, frag_BT_bf16(Bl[cur] + (wn + 16) * 32, 32, lane), acc1);
    if (have) {
      *(bf16x8*)(bl + (cur ^ 1) * PHASE) = cvt_bf16x8(nb0, nb1);
      wait_asynccnt0();
    }
    __syncthreads();
  }

  const int mb = m0 + wm + ((lane >> 4) << 3);
#pragma unroll
  for (int t = 0; t < 2; ++t) {
    const f32x8 acc = t ? acc1 : acc0;
    const int n = n0 + wn + (t << 4) + (lane & 15);
    const float bvv = bo[n];
#pragma unroll
    for (int r = 0; r < 8; ++r)
      out[(size_t)(mb + r) * DD + n] = acc[r] + bvv;
  }
}

// ---------------------------------------------------------------------------
// Kernel 4: attn_policy (B,1,N,N): ap[b,i,j] = pol[b,j] + (1-pol[b,j])*[i==j]
// ---------------------------------------------------------------------------
__global__ __launch_bounds__(256) void attn_policy_kernel(
    const float* __restrict__ policy, float* __restrict__ ap) {
  const size_t total4 = (size_t)BB * NN * NN / 4;
  size_t idx = (size_t)blockIdx.x * blockDim.x + threadIdx.x;
  const size_t stride = (size_t)gridDim.x * blockDim.x;
  for (; idx < total4; idx += stride) {
    const size_t base = idx * 4;
    const int j0 = (int)(base & (NN - 1));
    const size_t r = base >> 11;
    const int i = (int)(r & (NN - 1));
    const int b = (int)(r >> 11);
    const f32x4 p = *(const f32x4*)(policy + (size_t)b * NN + j0);
    f32x4 v;
#pragma unroll
    for (int e = 0; e < 4; ++e) v[e] = (j0 + e == i) ? 1.0f : p[e];
    *(f32x4*)(ap + base) = v;
  }
}

// ---------------------------------------------------------------------------
extern "C" void kernel_launch(void* const* d_in, const int* in_sizes, int n_in,
                              void* d_out, int out_size, void* d_ws, size_t ws_size,
                              hipStream_t stream) {
  (void)in_sizes; (void)n_in; (void)out_size; (void)ws_size;
  const float* hs  = (const float*)d_in[0];
  const float* pol = (const float*)d_in[1];
  const float* Wq  = (const float*)d_in[2];
  const float* bq  = (const float*)d_in[3];
  const float* Wk  = (const float*)d_in[4];
  const float* bk  = (const float*)d_in[5];
  const float* Wv  = (const float*)d_in[6];
  const float* bv  = (const float*)d_in[7];
  const float* Wo  = (const float*)d_in[8];
  const float* bo  = (const float*)d_in[9];

  float* out   = (float*)d_out;                       // B*N*D
  float* probs = out + (size_t)BN * DD;               // B*H*N*N
  float* apol  = probs + (size_t)BB * HH * NN * NN;   // B*N*N

  const size_t head_elems = (size_t)BB * HH * NN * DHH;   // 6291456
  __bf16* Qh    = (__bf16*)d_ws;
  __bf16* Kh    = Qh + head_elems;
  __bf16* Vt    = Kh + head_elems;                    // (B,H,DH,N)
  __bf16* ctx2d = Vt + head_elems;                    // B*N*D bf16

  // Allow 270KB dynamic LDS for the fused kernel (CDNA5: up to 320KB/WG)
  hipFuncSetAttribute(reinterpret_cast<const void*>(attn_fused_kernel),
                      hipFuncAttributeMaxDynamicSharedMemorySize, SM_BYTES);

  // QKV projections: 128 x 12 = 1536 blocks of 256 threads (64x64 tiles)
  proj_tiled_kernel<0><<<dim3(1536), dim3(256), 0, stream>>>(hs, Wq, bq, Qh, 0.125f);
  proj_tiled_kernel<0><<<dim3(1536), dim3(256), 0, stream>>>(hs, Wk, bk, Kh, 1.0f);
  proj_tiled_kernel<1><<<dim3(1536), dim3(256), 0, stream>>>(hs, Wv, bv, Vt, 1.0f);

  // Fused scores+softmax+ctx: 64 strips x 48 heads, 512 threads, 270KB LDS
  attn_fused_kernel<<<dim3(NN / STRIP, 1, BB * HH), dim3(512), SM_BYTES, stream>>>(
      Qh, Kh, Vt, pol, probs, ctx2d);

  // Output projection
  out_tiled_kernel<<<dim3(1536), dim3(256), 0, stream>>>(ctx2d, Wo, bo, out);

  // attn_policy fill (float4 grid-stride)
  attn_policy_kernel<<<dim3(4096), dim3(256), 0, stream>>>(pol, apol);
}
